// MultiHeadAttention_42279658061897
// MI455X (gfx1250) — matrix operand order repaired and checked by reading, not compiled, not run on hardware
//
#include <hip/hip_runtime.h>
#include <hip/hip_bf16.h>

// ---------------------------------------------------------------------------
// Types for CDNA5 WMMA (wave32): v_wmma_f32_16x16x32_bf16
// ---------------------------------------------------------------------------
typedef __bf16 bf16;
typedef __attribute__((ext_vector_type(16))) __bf16 v16bf;
typedef __attribute__((ext_vector_type(8)))  __bf16 v8bf;
typedef __attribute__((ext_vector_type(8)))  float  v8f;

union ABfrag { v16bf v; v8bf h[2]; };

__device__ __forceinline__ v8f zero_v8f() {
    v8f z;
#pragma unroll
    for (int i = 0; i < 8; ++i) z[i] = 0.0f;
    return z;
}

// f32 -> bf16, round-to-nearest-even, via bit ops
__device__ __forceinline__ bf16 f2bf(float x) {
    unsigned u = __builtin_bit_cast(unsigned, x);
    unsigned r = (u + 0x7FFFu + ((u >> 16) & 1u)) >> 16;
    unsigned short s = (unsigned short)r;
    return __builtin_bit_cast(bf16, s);
}
__device__ __forceinline__ float bf2f(bf16 x) {
    unsigned short s = __builtin_bit_cast(unsigned short, x);
    unsigned u = ((unsigned)s) << 16;
    return __builtin_bit_cast(float, u);
}

__device__ __forceinline__ v8f wmma_bf16(v16bf a, v16bf b, v8f c) {
    return __builtin_amdgcn_wmma_f32_16x16x32_bf16(
        false, a, false, b, (short)0, c, false, false);
}

// ---------------------------------------------------------------------------
// CDNA5 async global->LDS copy (ASYNCcnt-tracked), 16 bytes per lane.
// ---------------------------------------------------------------------------
__device__ __forceinline__ void async_copy16(const bf16* g, bf16* l) {
    unsigned loff = (unsigned)(uintptr_t)l;
    asm volatile("global_load_async_to_lds_b128 %0, %1, off"
                 :: "v"(loff), "v"(g) : "memory");
}
#define WAIT_ASYNC_LE_8() asm volatile("s_wait_asynccnt 0x8" ::: "memory")
#define WAIT_ASYNC_LE_0() asm volatile("s_wait_asynccnt 0x0" ::: "memory")

// A fragment from global row-major A[M,K] at (row_base, k0).
__device__ __forceinline__ v16bf load_afrag_g(const bf16* __restrict__ A, int lda,
                                              int row_base, int k0, int lane) {
    int m  = row_base + (lane & 15);
    int lo = (lane < 16) ? 0 : 8;
    ABfrag f;
    const bf16* p = A + (size_t)m * lda + (k0 + lo);
    f.h[0] = *(const v8bf*)(p);
    f.h[1] = *(const v8bf*)(p + 16);
    return f.v;
}

// ---------------------------------------------------------------------------
// Elementwise cast f32 -> bf16
// ---------------------------------------------------------------------------
__global__ void cast_f32_bf16(const float* __restrict__ in, bf16* __restrict__ out, size_t n) {
    size_t i = (size_t)blockIdx.x * blockDim.x + threadIdx.x;
    size_t stride = (size_t)gridDim.x * blockDim.x;
    for (; i < n; i += stride) out[i] = f2bf(in[i]);
}

// Transpose + cast: W[K,N] f32 -> WT[N,K] bf16
__global__ void transpose_cast(const float* __restrict__ W, bf16* __restrict__ WT,
                               int Kd, int Nd) {
    int idx = blockIdx.x * blockDim.x + threadIdx.x;
    if (idx >= Kd * Nd) return;
    int n = idx / Kd, k = idx % Kd;
    WT[idx] = f2bf(W[(size_t)k * Nd + n]);
}

// Per-batch head transpose for V: Vp[32768,64] bf16 -> Vt[16][64][2048] bf16
__global__ void vhead_transpose(const bf16* __restrict__ Vp, bf16* __restrict__ Vt) {
    int idx = blockIdx.x * blockDim.x + threadIdx.x;   // 2M
    if (idx >= 16 * 64 * 2048) return;
    int t = idx & 2047;
    int e = (idx >> 11) & 63;
    int g = idx >> 17;
    Vt[idx] = Vp[((size_t)(g * 2048 + t)) * 64 + e];
}

// ---------------------------------------------------------------------------
// bf16 WMMA GEMM, async-LDS double buffered, K-step 64:
//   C[M,N](f32) = A[M,K](bf16) * BT[N,K](bf16)^T + bias
// Block = 8 waves; block tile 128x128. Panels 128x64 staged one step ahead via
// global_load_async_to_lds_b128 (8 per wave per stage). LDS rows padded to 72
// bf16 (144 B): 36*l mod 64 is a permutation over 16 lanes -> conflict-free
// b128 fragment reads. B fragments register-pipelined across the j-loop.
// ---------------------------------------------------------------------------
#define TK  64
#define LDP 72   // padded row length in bf16 elements (64 data + 8 pad)

__global__ __launch_bounds__(256) void gemm_bf16_wmma(
    const bf16* __restrict__ A, const bf16* __restrict__ BT,
    const float* __restrict__ bias,
    float* __restrict__ C, bf16* __restrict__ Cb,
    int M, int N, int K) {
    __shared__ __align__(16) bf16 smA[2][128 * LDP];   // 2 x 18 KB
    __shared__ __align__(16) bf16 smB[2][128 * LDP];   // 2 x 18 KB

    int tid  = threadIdx.x;
    int lane = tid & 31;
    int wave = tid >> 5;
    int mBase = blockIdx.y * 128;
    int nBase = blockIdx.x * 128;

    // Staging map: thread -> (rows r0+{0,32,64,96}, 16B chunk c0), both panels.
    int r0 = tid >> 3;          // 0..31
    int c0 = (tid & 7) * 8;     // element offset of 16B chunk within 64-wide row

    auto stage = [&](int buf, int k0) {
#pragma unroll
        for (int rr = 0; rr < 128; rr += 32) {
            const bf16* Ag = A  + (size_t)(mBase + r0 + rr) * K + k0 + c0;
            async_copy16(Ag, &smA[buf][(r0 + rr) * LDP + c0]);
        }
#pragma unroll
        for (int rr = 0; rr < 128; rr += 32) {
            const bf16* Bg = BT + (size_t)(nBase + r0 + rr) * K + k0 + c0;
            async_copy16(Bg, &smB[buf][(r0 + rr) * LDP + c0]);
        }
    };

    v8f acc[8];
#pragma unroll
    for (int j = 0; j < 8; ++j) acc[j] = zero_v8f();

    int lo  = (lane < 16) ? 0 : 8;    // A-fragment K-half select
    int klo = (lane < 16) ? 0 : 16;   // unused; B fragments select their K-half with lo
    int nl  = lane & 15;
    int mA  = wave * 16 + nl;         // A row within block tile for this lane

    stage(0, 0);
    int cur = 0;
    for (int k0 = 0; k0 < K; k0 += TK) {
        bool pre = (k0 + TK < K);
        if (pre) stage(cur ^ 1, k0 + TK);           // prefetch next K-block
        if (pre) { WAIT_ASYNC_LE_8(); } else { WAIT_ASYNC_LE_0(); }
        __syncthreads();                             // cur buffer visible

        // A fragments for both 32-deep halves of this K-block
        ABfrag a0, a1;
        const bf16* pa = &smA[cur][mA * LDP + lo];
        a0.h[0] = *(const v8bf*)(pa);       a0.h[1] = *(const v8bf*)(pa + 16);
        a1.h[0] = *(const v8bf*)(pa + 32);  a1.h[1] = *(const v8bf*)(pa + 48);

        // B fragments register-pipelined one j ahead
        ABfrag b0, b1, n0f, n1f;
        {
            const bf16* pb = &smB[cur][nl * LDP + lo];
            b0.h[0] = *(const v8bf*)(pb);       b0.h[1] = *(const v8bf*)(pb + 16);
            b1.h[0] = *(const v8bf*)(pb + 32);  b1.h[1] = *(const v8bf*)(pb + 48);
        }
#pragma unroll
        for (int j = 0; j < 8; ++j) {
            if (j < 7) {
                const bf16* pb = &smB[cur][((j + 1) * 16 + nl) * LDP + lo];
                n0f.h[0] = *(const v8bf*)(pb);       n0f.h[1] = *(const v8bf*)(pb + 16);
                n1f.h[0] = *(const v8bf*)(pb + 32);  n1f.h[1] = *(const v8bf*)(pb + 48);
            }
            acc[j] = wmma_bf16(a0.v, b0.v, acc[j]);
            acc[j] = wmma_bf16(a1.v, b1.v, acc[j]);
            b0 = n0f; b1 = n1f;
        }
        __syncthreads();                             // reads done before overwrite
        cur ^= 1;
    }

    int hi = lane >> 4;
#pragma unroll
    for (int j = 0; j < 8; ++j) {
        int n = nBase + j * 16 + nl;
        float bv = bias ? bias[n] : 0.0f;
#pragma unroll
        for (int r = 0; r < 8; ++r) {
            int m = mBase + wave * 16 + r + 8 * hi;
            float v = acc[j][r] + bv;
            if (C)  C[(size_t)m * N + n] = v;
            if (Cb) Cb[(size_t)m * N + n] = f2bf(v);
        }
    }
}

// ---------------------------------------------------------------------------
// Fused attention for one (batch-head g, 32-row q-tile):
//   scores = q(32x64) @ K^T(64x2048) * scale -> LDS bf16 (128 KB)
//   softmax rows in-place (row slice held in registers, f32 statistics)
//   Z(32x64) = P(32x2048) @ V(2048x64)       (Vt[g][64][2048] pre-transposed)
// ---------------------------------------------------------------------------
__global__ __launch_bounds__(256) void attn_kernel(
    const bf16* __restrict__ Qp, const bf16* __restrict__ Kp,
    const bf16* __restrict__ Vt, bf16* __restrict__ Zb, float scale) {
    __shared__ __align__(16) bf16 Sb[32 * 2048];   // 128 KB of 320 KB WGP LDS

    int lane = threadIdx.x & 31;
    int wave = threadIdx.x >> 5;
    int qt = blockIdx.x;        // 0..63
    int g  = blockIdx.y;        // 0..15
    int nl  = lane & 15;
    int lo  = (lane < 16) ? 0 : 8;
    int klo = (lane < 16) ? 0 : 16;

    const bf16* Qg = Qp + (size_t)g * 2048 * 64;
    const bf16* Kg = Kp + (size_t)g * 2048 * 64;
    int qbase = qt * 32;

    v16bf afrag[2][2];
#pragma unroll
    for (int qs = 0; qs < 2; ++qs)
#pragma unroll
        for (int kk = 0; kk < 2; ++kk)
            afrag[qs][kk] = load_afrag_g(Qg, 64, qbase + qs * 16, kk * 32, lane);

    // ---- Phase 1: scores -> Sb (bf16, scaled); B fragments pipelined ----
    {
        auto loadB = [&](ABfrag* d, int t) {
            int nt = t & 127;
            const bf16* p = Kg + (size_t)(nt * 16 + nl) * 64 + lo;
#pragma unroll
            for (int kk = 0; kk < 2; ++kk) {
                d[kk].h[0] = *(const v8bf*)(p + kk * 32);
                d[kk].h[1] = *(const v8bf*)(p + kk * 32 + 16);
            }
        };
        ABfrag bc[2], bn[2];
        loadB(bc, wave);
        for (int t = wave; t < 256; t += 8) {
            if (t + 8 < 256) loadB(bn, t + 8);
            int qs = t >> 7;
            int nt = t & 127;
            v8f acc = zero_v8f();
            acc = wmma_bf16(afrag[qs][0], bc[0].v, acc);
            acc = wmma_bf16(afrag[qs][1], bc[1].v, acc);
            int n  = nt * 16 + nl;
            int hi = lane >> 4;
#pragma unroll
            for (int r = 0; r < 8; ++r)
                Sb[(size_t)(qs * 16 + r + 8 * hi) * 2048 + n] = f2bf(acc[r] * scale);
            bc[0] = bn[0]; bc[1] = bn[1];
        }
    }
    __syncthreads();

    // ---- Phase 2: softmax; lane holds its contiguous 64-elem slice in regs ----
#pragma unroll
    for (int rr = 0; rr < 4; ++rr) {
        int row = wave * 4 + rr;
        v8bf* slice = (v8bf*)(Sb + (size_t)row * 2048 + lane * 64);
        v8bf vals[8];
#pragma unroll
        for (int i = 0; i < 8; ++i) vals[i] = slice[i];     // 8 x ds_load_b128
        float mx = -1e30f;
#pragma unroll
        for (int i = 0; i < 8; ++i)
#pragma unroll
            for (int j = 0; j < 8; ++j) mx = fmaxf(mx, bf2f(vals[i][j]));
#pragma unroll
        for (int off = 16; off >= 1; off >>= 1) mx = fmaxf(mx, __shfl_xor(mx, off, 32));
        float e[8][8];
        float sum = 0.0f;
#pragma unroll
        for (int i = 0; i < 8; ++i)
#pragma unroll
            for (int j = 0; j < 8; ++j) { e[i][j] = __expf(bf2f(vals[i][j]) - mx); sum += e[i][j]; }
#pragma unroll
        for (int off = 16; off >= 1; off >>= 1) sum += __shfl_xor(sum, off, 32);
        float inv = 1.0f / sum;
#pragma unroll
        for (int i = 0; i < 8; ++i) {
            v8bf o;
#pragma unroll
            for (int j = 0; j < 8; ++j) o[j] = f2bf(e[i][j] * inv);
            slice[i] = o;                                   // 8 x ds_store_b128
        }
    }
    __syncthreads();

    // ---- Phase 3: Z tile per wave; A(LDS)/B(global) pipelined over K ----
    {
        int qs = wave >> 2;
        int n0 = (wave & 3) * 16;
        const bf16* Vg = Vt + (size_t)g * 64 * 2048;
        int m  = qs * 16 + nl;
        const bf16* prow = Sb + (size_t)m * 2048 + lo;
        const bf16* vrow = Vg + (size_t)(n0 + nl) * 2048 + lo;

        ABfrag ac, bc, an, bn;
        ac.h[0] = *(const v8bf*)(prow);      ac.h[1] = *(const v8bf*)(prow + 16);
        bc.h[0] = *(const v8bf*)(vrow);      bc.h[1] = *(const v8bf*)(vrow + 16);
        v8f acc = zero_v8f();
#pragma unroll 4
        for (int k0 = 0; k0 < 2048; k0 += 32) {
            if (k0 + 32 < 2048) {
                an.h[0] = *(const v8bf*)(prow + k0 + 32);  an.h[1] = *(const v8bf*)(prow + k0 + 48);
                bn.h[0] = *(const v8bf*)(vrow + k0 + 32);  bn.h[1] = *(const v8bf*)(vrow + k0 + 48);
            }
            acc = wmma_bf16(ac.v, bc.v, acc);
            ac = an; bc = bn;
        }
        int hi = lane >> 4;
        int n  = n0 + nl;
#pragma unroll
        for (int r = 0; r < 8; ++r) {
            size_t row = (size_t)g * 2048 + qbase + qs * 16 + r + 8 * hi;
            Zb[row * 64 + n] = f2bf(acc[r]);
        }
    }
}

// ---------------------------------------------------------------------------
// out = LayerNorm(X + Y) * gamma + beta ; one wave per row of 512 (wave32)
// ---------------------------------------------------------------------------
__global__ __launch_bounds__(256) void residual_ln_kernel(
    const float* __restrict__ X, const float* __restrict__ Y,
    const float* __restrict__ gamma, const float* __restrict__ beta,
    float* __restrict__ outf, bf16* __restrict__ outb, int Mrows) {
    int lane = threadIdx.x & 31;
    int wave = threadIdx.x >> 5;
    int row = blockIdx.x * 8 + wave;
    if (row >= Mrows) return;
    const float* xr = X + (size_t)row * 512;
    const float* yr = Y + (size_t)row * 512;
    float v[16];
    float s = 0.0f;
#pragma unroll
    for (int i = 0; i < 16; ++i) { v[i] = xr[lane + i * 32] + yr[lane + i * 32]; s += v[i]; }
#pragma unroll
    for (int off = 16; off >= 1; off >>= 1) s += __shfl_xor(s, off, 32);
    float mu = s * (1.0f / 512.0f);
    float q = 0.0f;
#pragma unroll
    for (int i = 0; i < 16; ++i) { float d = v[i] - mu; q += d * d; }
#pragma unroll
    for (int off = 16; off >= 1; off >>= 1) q += __shfl_xor(q, off, 32);
    float rstd = rsqrtf(q * (1.0f / 512.0f) + 1e-5f);
#pragma unroll
    for (int i = 0; i < 16; ++i) {
        int c = lane + i * 32;
        float o = (v[i] - mu) * rstd * gamma[c] + beta[c];
        outf[(size_t)row * 512 + c] = o;
        if (outb) outb[(size_t)row * 512 + c] = f2bf(o);
    }
}

// ---------------------------------------------------------------------------
// Host launcher
// ---------------------------------------------------------------------------
extern "C" void kernel_launch(void* const* d_in, const int* in_sizes, int n_in,
                              void* d_out, int out_size, void* d_ws, size_t ws_size,
                              hipStream_t stream) {
    (void)in_sizes; (void)n_in; (void)out_size; (void)ws_size;
    const int Mrows = 4096;            // B*S
    const int D = 512, FF = 2048;
    const float scale = 0.044194173824159216f;   // 1/sqrt(512)

    const float* Qin   = (const float*)d_in[0];
    const float* Kin   = (const float*)d_in[1];
    const float* Vin   = (const float*)d_in[2];
    /* d_in[3] = mask: unused by the reference math */
    const float* Wq    = (const float*)d_in[4];
    const float* Wk    = (const float*)d_in[5];
    const float* Wv    = (const float*)d_in[6];
    const float* Wo    = (const float*)d_in[7];
    const float* bo    = (const float*)d_in[8];
    const float* gamma = (const float*)d_in[9];
    const float* beta  = (const float*)d_in[10];
    const float* W1    = (const float*)d_in[11];
    const float* b1    = (const float*)d_in[12];
    const float* W2    = (const float*)d_in[13];
    const float* b2    = (const float*)d_in[14];
    float* out = (float*)d_out;

    size_t off = 0;
    auto carve = [&](size_t bytes) {
        void* p = (char*)d_ws + off;
        off += (bytes + 255) & ~(size_t)255;
        return p;
    };
    const size_t nMD  = (size_t)Mrows * D;     // 2M
    const size_t nMF  = (size_t)Mrows * FF;    // 8M
    bf16* Qib = (bf16*)carve(nMD * 2);
    bf16* Kib = (bf16*)carve(nMD * 2);
    bf16* Vib = (bf16*)carve(nMD * 2);
    bf16* WqT = (bf16*)carve((size_t)D * D * 2);
    bf16* WkT = (bf16*)carve((size_t)D * D * 2);
    bf16* WvT = (bf16*)carve((size_t)D * D * 2);
    bf16* WoT = (bf16*)carve((size_t)D * D * 2);
    bf16* W1T = (bf16*)carve((size_t)D * FF * 2);
    bf16* W2T = (bf16*)carve((size_t)D * FF * 2);
    bf16* Qpb = (bf16*)carve(nMD * 2);
    bf16* Kpb = (bf16*)carve(nMD * 2);
    bf16* Vpb = (bf16*)carve(nMD * 2);
    bf16* Vt  = (bf16*)carve(nMD * 2);
    bf16* Zb  = (bf16*)carve(nMD * 2);
    float* O  = (float*)carve(nMD * 4);
    float* Zn = (float*)carve(nMD * 4);
    bf16* Znb = (bf16*)carve(nMD * 2);
    bf16* H1b = (bf16*)carve(nMF * 2);
    float* F2 = (float*)carve(nMD * 4);

    // 1) cast activations
    cast_f32_bf16<<<2048, 256, 0, stream>>>(Qin, Qib, nMD);
    cast_f32_bf16<<<2048, 256, 0, stream>>>(Kin, Kib, nMD);
    cast_f32_bf16<<<2048, 256, 0, stream>>>(Vin, Vib, nMD);

    // 2) transpose+cast weights: W[K,N] -> WT[N,K]
    transpose_cast<<<(D * D + 255) / 256, 256, 0, stream>>>(Wq, WqT, D, D);
    transpose_cast<<<(D * D + 255) / 256, 256, 0, stream>>>(Wk, WkT, D, D);
    transpose_cast<<<(D * D + 255) / 256, 256, 0, stream>>>(Wv, WvT, D, D);
    transpose_cast<<<(D * D + 255) / 256, 256, 0, stream>>>(Wo, WoT, D, D);
    transpose_cast<<<(D * FF + 255) / 256, 256, 0, stream>>>(W1, W1T, D, FF);
    transpose_cast<<<(D * FF + 255) / 256, 256, 0, stream>>>(W2, W2T, FF, D);

    // 3) QKV projections
    {
        dim3 grid(D / 128, Mrows / 128);
        gemm_bf16_wmma<<<grid, 256, 0, stream>>>(Qib, WqT, nullptr, nullptr, Qpb, Mrows, D, D);
        gemm_bf16_wmma<<<grid, 256, 0, stream>>>(Kib, WkT, nullptr, nullptr, Kpb, Mrows, D, D);
        gemm_bf16_wmma<<<grid, 256, 0, stream>>>(Vib, WvT, nullptr, nullptr, Vpb, Mrows, D, D);
    }

    // 4) V head-transpose
    vhead_transpose<<<(16 * 64 * 2048 + 255) / 256, 256, 0, stream>>>(Vpb, Vt);

    // 5) fused attention
    attn_kernel<<<dim3(64, 16), 256, 0, stream>>>(Qpb, Kpb, Vt, Zb, scale);

    // 6) output projection + bo
    gemm_bf16_wmma<<<dim3(D / 128, Mrows / 128), 256, 0, stream>>>(
        Zb, WoT, bo, O, nullptr, Mrows, D, D);

    // 7) Zn = LN(Vin + O)
    residual_ln_kernel<<<Mrows / 8, 256, 0, stream>>>(Vin, O, gamma, beta, Zn, Znb, Mrows);

    // 8) FFN up
    gemm_bf16_wmma<<<dim3(FF / 128, Mrows / 128), 256, 0, stream>>>(
        Znb, W1T, b1, nullptr, H1b, Mrows, FF, D);

    // 9) FFN down
    gemm_bf16_wmma<<<dim3(D / 128, Mrows / 128), 256, 0, stream>>>(
        H1b, W2T, b2, F2, nullptr, Mrows, D, FF);

    // 10) out = LN(Zn + F2)
    residual_ln_kernel<<<Mrows / 8, 256, 0, stream>>>(Zn, F2, gamma, beta, out, nullptr, Mrows);
}
